// Kplanes_69140383531572
// MI455X (gfx1250) — compile-verified, matrix operand order checked by reading, not run on hardware
//
#include <hip/hip_runtime.h>
#include <hip/hip_bf16.h>

typedef __attribute__((ext_vector_type(16))) _Float16 v16h;
typedef __attribute__((ext_vector_type(8)))  float    v8f;

namespace {
constexpr int kC  = 32;          // plane channels
constexpr int kN  = 512;         // plane H = W
constexpr int kHW = kN * kN;
constexpr int kP  = 2000000;     // points
constexpr int kH1 = 128;         // hidden width
static_assert(kP % 128 == 0, "grid sizing");
}

// ---------------------------------------------------------------------------
// Kernel 1: [C,H,W] -> [H,W,C] transpose of the three planes into workspace.
// LDS-tiled so both global reads and writes are 128B-coalesced.
// grid.x = 3 planes * 512 rows * 16 x-blocks, block = 256 threads.
// ---------------------------------------------------------------------------
__global__ void kplanes_transpose(const float* __restrict__ p0,
                                  const float* __restrict__ p1,
                                  const float* __restrict__ p2,
                                  float* __restrict__ ws) {
  __shared__ float tile[kC][33];   // pad to kill bank conflicts
  const int bid = blockIdx.x;
  const int xb  = bid & 15;              // 16 blocks of 32 columns
  const int y   = (bid >> 4) & (kN - 1); // row
  const int pl  = bid >> 13;             // plane (8192 = 512*16)
  const float* __restrict__ src = (pl == 0) ? p0 : (pl == 1) ? p1 : p2;
  float* __restrict__ dst = ws + (size_t)pl * (size_t)(kC * kHW);

  const int t = threadIdx.x;
#pragma unroll
  for (int i = 0; i < 4; ++i) {          // read: lanes sweep x (coalesced)
    const int c = i * 8 + (t >> 5);
    const int x = xb * 32 + (t & 31);
    tile[c][t & 31] = src[(size_t)c * kHW + (size_t)y * kN + x];
  }
  __syncthreads();
#pragma unroll
  for (int i = 0; i < 4; ++i) {          // write: lanes sweep c (coalesced)
    const int c  = t & 31;
    const int xi = i * 8 + (t >> 5);
    const int x  = xb * 32 + xi;
    dst[((size_t)(y * kN + x)) * kC + c] = tile[c][xi];
  }
}

// ---------------------------------------------------------------------------
// Bilinear sample of 16 channels for one point, in WMMA A-fragment element
// order: element e -> channel ch(e) = (e<8 ? cb0+e : cb0+16+(e-8)).
//   TR=true : plane stored [H,W,C] (channel-innermost, float4 corner loads)
//   TR=false: plane stored [C,H,W] (per-channel scalar gathers)
// align_corners=True; the x0/y0 clamp-to-(dim-2) trick reproduces the
// reference's zero-padded corner exactly (OOB corner always has weight 0).
// ---------------------------------------------------------------------------
template <bool TR>
__device__ inline void sample_plane(const float* __restrict__ base, float gx,
                                    float gy, int cb0, float v[16]) {
  float x = (gx + 1.0f) * 0.5f * (float)(kN - 1);
  float y = (gy + 1.0f) * 0.5f * (float)(kN - 1);
  x = fminf(fmaxf(x, 0.0f), (float)(kN - 1));
  y = fminf(fmaxf(y, 0.0f), (float)(kN - 1));
  int x0 = (int)floorf(x); if (x0 > kN - 2) x0 = kN - 2;
  int y0 = (int)floorf(y); if (y0 > kN - 2) y0 = kN - 2;
  const float wx = x - (float)x0, wy = y - (float)y0;
  const float w00 = (1.0f - wx) * (1.0f - wy);
  const float w10 = wx * (1.0f - wy);
  const float w01 = (1.0f - wx) * wy;
  const float w11 = wx * wy;

  if (TR) {
    const float* __restrict__ c00 = base + (size_t)(y0 * kN + x0) * kC;
    const float* __restrict__ c10 = c00 + kC;
    const float* __restrict__ c01 = c00 + (size_t)kN * kC;
    const float* __restrict__ c11 = c01 + kC;
#pragma unroll
    for (int g = 0; g < 2; ++g) {
      const int cb = cb0 + g * 16;
#pragma unroll
      for (int q = 0; q < 2; ++q) {
        const int o = cb + 4 * q;
        const float4 a = *(const float4*)(c00 + o);
        const float4 b = *(const float4*)(c10 + o);
        const float4 c = *(const float4*)(c01 + o);
        const float4 d = *(const float4*)(c11 + o);
        const int e = g * 8 + 4 * q;
        v[e + 0] = w00 * a.x + w10 * b.x + w01 * c.x + w11 * d.x;
        v[e + 1] = w00 * a.y + w10 * b.y + w01 * c.y + w11 * d.y;
        v[e + 2] = w00 * a.z + w10 * b.z + w01 * c.z + w11 * d.z;
        v[e + 3] = w00 * a.w + w10 * b.w + w01 * c.w + w11 * d.w;
      }
    }
  } else {
#pragma unroll
    for (int e = 0; e < 16; ++e) {
      const int ch = (e < 8) ? (cb0 + e) : (cb0 + 8 + e);
      const float* __restrict__ rp =
          base + (size_t)ch * kHW + (size_t)y0 * kN + x0;
      const float a = rp[0], b = rp[1];
      const float c = rp[kN], d = rp[kN + 1];
      v[e] = w00 * a + w10 * b + w01 * c + w11 * d;
    }
  }
}

// ---------------------------------------------------------------------------
// Kernel 2: gather + fused MLP. Block = 256 (8 wave32), wave owns 16 points.
// A-fragment: lanes l / l+16 hold channel halves {0-7,16-23} / {8-15,24-31}
// of point l. 8x v_wmma_f32_16x16x32_f16 per wave (K=32 == channel count).
// W1 is staged in LDS pre-converted to f16 and pre-swizzled into exact
// B-fragment order (one 32B LDS read per tile). WMMA uses inline SRC2=0;
// bias+ReLU+W2-dot are fused on the accumulator (add-first so maxnum needs
// no canonicalize), keeping VGPR pressure and post-WMMA latency low.
// ---------------------------------------------------------------------------
template <bool TR>
__global__ void __launch_bounds__(256)
kplanes_main(const float* __restrict__ coords, const float* __restrict__ p0,
             const float* __restrict__ p1, const float* __restrict__ p2,
             const float* __restrict__ W1, const float* __restrict__ b1,
             const float* __restrict__ W2, const float* __restrict__ b2,
             float* __restrict__ out, const float* __restrict__ wsT) {
  // W1 in B-fragment order: [tile(8)][lane(32)][elem(16)] halves = 8 KB.
  __shared__ _Float16 sW1h[8 * 32 * 16];
  __shared__ float sb1[kH1];
  __shared__ float sW2[kH1];

  const int t = threadIdx.x;
  // Pre-swizzle + f16-convert W1 (one pass, L2-hot 16 KB source).
  for (int i = t; i < 8 * 32 * 16; i += 256) {
    const int t8 = i >> 9;         // tile index
    const int ln = (i >> 4) & 31;  // destination lane
    const int e  = i & 15;         // fragment element
    const int cb = (ln >> 4) * 8;  // channel half base for that lane
    const int k  = (e < 8) ? (cb + e) : (cb + 8 + e);
    const int cl = t8 * 16 + (ln & 15);
    sW1h[i] = (_Float16)W1[k * kH1 + cl];
  }
  if (t < kH1) { sb1[t] = b1[t]; sW2[t] = W2[t]; }
  __syncthreads();
  const float b2v = b2[0];

  const int wave = t >> 5;
  const int lane = t & 31;
  const int nl   = lane & 15;      // row-of-tile for A / column for B,D
  const int half = lane >> 4;      // channel half selector
  const int cb0  = half * 8;       // base channel of this lane's halves
  const int wb   = (blockIdx.x * 8 + wave) * 16;  // wave's first point
  const int pt   = wb + nl;        // point this lane gathers for

  // --- coordinates ([1,P,3] flat) ---
  const float cx = coords[3 * pt + 0];
  const float cy = coords[3 * pt + 1];
  const float cz = coords[3 * pt + 2];

  // --- tri-plane bilinear gather (channel order = A-fragment order) ---
  const float* __restrict__ b0  = TR ? (wsT + (size_t)0 * kC * kHW) : p0;
  const float* __restrict__ bp1 = TR ? (wsT + (size_t)1 * kC * kHW) : p1;
  const float* __restrict__ bp2 = TR ? (wsT + (size_t)2 * kC * kHW) : p2;
  float f0[16], f1[16], f2[16];
  sample_plane<TR>(b0,  cx, cy, cb0, f0);   // xy plane
  sample_plane<TR>(bp1, cy, cz, cb0, f1);   // yz plane
  sample_plane<TR>(bp2, cx, cz, cb0, f2);   // xz plane

  v16h afrag;
#pragma unroll
  for (int e = 0; e < 16; ++e) {
    float f = f0[e] * f1[e] * f2[e];
    f = fmaxf(f, 0.0f);                     // h = relu(feats)
    afrag[e] = (_Float16)f;
  }

  // --- MLP: 8 WMMA tiles over hidden dim, fused bias/ReLU/W2-dot ---
  const _Float16* __restrict__ myB = &sW1h[lane * 16];
  float s[8];
#pragma unroll
  for (int r = 0; r < 8; ++r) s[r] = 0.0f;

#pragma unroll
  for (int t8 = 0; t8 < 8; ++t8) {
    const int col = t8 * 16 + nl;           // this lane's hidden column
    const v16h bfrag = *(const v16h*)(myB + t8 * (32 * 16));
    const float b1v = sb1[col];
    const float w2v = sW2[col];
    v8f acc = {};                            // SRC2 = inline 0, no setup movs
    acc = __builtin_amdgcn_wmma_f32_16x16x32_f16(
        false, afrag, false, bfrag, (short)0, acc, false, false);
#pragma unroll
    for (int r = 0; r < 8; ++r) {
      const float h = fmaxf(acc[r] + b1v, 0.0f);   // relu(h@W1 + b1)
      s[r] = fmaf(h, w2v, s[r]);                   // partial of h@W2
    }
  }

  // --- reduce each row's partial across the 16 lanes of its half-wave ---
  // D layout: lanes 0-15 hold rows 0-7, lanes 16-31 hold rows 8-15.
#pragma unroll
  for (int r = 0; r < 8; ++r) {
    float v = s[r];
    v += __shfl_xor(v, 1, 32);
    v += __shfl_xor(v, 2, 32);
    v += __shfl_xor(v, 4, 32);
    v += __shfl_xor(v, 8, 32);
    s[r] = v;
  }
  if (nl < 8) {
    out[wb + half * 8 + nl] = s[nl] + b2v;
  }
}

// ---------------------------------------------------------------------------
// Launch: transpose planes into workspace if it fits (96 MiB), else gather
// directly from [C,H,W]. Deterministic: path depends only on ws_size.
// ---------------------------------------------------------------------------
extern "C" void kernel_launch(void* const* d_in, const int* in_sizes, int n_in,
                              void* d_out, int out_size, void* d_ws,
                              size_t ws_size, hipStream_t stream) {
  (void)in_sizes; (void)n_in; (void)out_size;
  const float* coords = (const float*)d_in[0];
  const float* p0 = (const float*)d_in[1];
  const float* p1 = (const float*)d_in[2];
  const float* p2 = (const float*)d_in[3];
  const float* W1 = (const float*)d_in[4];
  const float* b1 = (const float*)d_in[5];
  const float* W2 = (const float*)d_in[6];
  const float* b2 = (const float*)d_in[7];
  float* out = (float*)d_out;

  const size_t needWs = (size_t)3 * kC * kHW * sizeof(float);  // ~96 MiB
  const int grid = kP / 128;                                   // 15625

  if (d_ws != nullptr && ws_size >= needWs) {
    float* wsT = (float*)d_ws;
    kplanes_transpose<<<3 * kN * 16, 256, 0, stream>>>(p0, p1, p2, wsT);
    kplanes_main<true><<<grid, 256, 0, stream>>>(coords, p0, p1, p2, W1, b1,
                                                 W2, b2, out, wsT);
  } else {
    kplanes_main<false><<<grid, 256, 0, stream>>>(coords, p0, p1, p2, W1, b1,
                                                  W2, b2, out, nullptr);
  }
}
